// VMGAE_11433202942400
// MI455X (gfx1250) — matrix operand order repaired and checked by reading, not compiled, not run on hardware
//
#include <hip/hip_runtime.h>
#include <hip/hip_bf16.h>
#include <math.h>

#define N_NODES 8192
#define F_DIM   512
#define H_DIM   256
#define O_DIM   128
#define C_DIM   64
#define E_EDGES 262144

typedef __attribute__((ext_vector_type(16))) __bf16 v16bf;
typedef __attribute__((ext_vector_type(8)))  __bf16 v8bf;
typedef __attribute__((ext_vector_type(8)))  float  v8f;

// ---------- helpers ----------
__device__ __forceinline__ __bf16 f2bf(float f) {
    unsigned u = __float_as_uint(f);
    unsigned r = (u + 0x7FFFu + ((u >> 16) & 1u)) >> 16;   // round-to-nearest-even
    unsigned short s = (unsigned short)r;
    union { unsigned short u16; __bf16 b; } cvt;
    cvt.u16 = s;
    return cvt.b;
}

// ---------- elementwise kernels ----------
__global__ void fill_f32(float* p, float v, int n) {
    int i = blockIdx.x * blockDim.x + threadIdx.x;
    if (i < n) p[i] = v;
}

__global__ void deg_count(const long long* dst, float* deg, int e) {
    int i = blockIdx.x * blockDim.x + threadIdx.x;
    if (i < e) atomicAdd(&deg[(int)dst[i]], 1.0f);
}

__global__ void rsqrt_k(const float* deg, float* dinv, int n) {
    int i = blockIdx.x * blockDim.x + threadIdx.x;
    if (i < n) dinv[i] = rsqrtf(deg[i]);
}

__global__ void cvt_bf16(const float* a, __bf16* b, int n) {
    int i = blockIdx.x * blockDim.x + threadIdx.x;
    if (i < n) b[i] = f2bf(a[i]);
}

// w[K][Nn] (row-major) -> wt[Nn][K] bf16 (K-contiguous per output column)
__global__ void cvt_t_bf16(const float* w, __bf16* wt, int K, int Nn) {
    int i = blockIdx.x * blockDim.x + threadIdx.x;
    if (i < K * Nn) {
        int k = i / Nn, n = i % Nn;
        wt[(size_t)n * K + k] = f2bf(w[i]);
    }
}

// hout[dst] += hin[src] * dinv[src]*dinv[dst], vectorized over 4 channels
__global__ void scatter_add(const long long* src, const long long* dst, const float* dinv,
                            const float* hin, float* hout, int Hc, int e) {
    int i = blockIdx.x * blockDim.x + threadIdx.x;
    int cpe = Hc >> 2;                 // float4 chunks per edge
    if (i >= e * cpe) return;
    int ei = i / cpe;
    int c4 = (i - ei * cpe) << 2;
    int s = (int)src[ei], d = (int)dst[ei];
    float nrm = dinv[s] * dinv[d];
    const float4 hv = *(const float4*)(hin + (size_t)s * Hc + c4);
    float* o = hout + (size_t)d * Hc + c4;
    atomicAdd(o + 0, hv.x * nrm);
    atomicAdd(o + 1, hv.y * nrm);
    atomicAdd(o + 2, hv.z * nrm);
    atomicAdd(o + 3, hv.w * nrm);
}

// out[i] = (relu?) agg[i] + hw[i]*dinv[node]^2 + bias[c]
__global__ void finalize_k(const float* agg, const float* hw, const float* dinv,
                           const float* bias, float* outp, int Hc, int total, int do_relu) {
    int i = blockIdx.x * blockDim.x + threadIdx.x;
    if (i >= total) return;
    int node = i / Hc, c = i - node * Hc;
    float di = dinv[node];
    float v = agg[i] + hw[i] * di * di + bias[c];
    if (do_relu) v = fmaxf(v, 0.0f);
    outp[i] = v;
}

// zr = mu + eps * exp(0.5*logvar), emitted as bf16 (feeds final WMMA GEMM)
__global__ void reparam_k(const float* mu, const float* lv, const float* eps,
                          __bf16* zrbf, int n) {
    int i = blockIdx.x * blockDim.x + threadIdx.x;
    if (i < n) {
        float v = mu[i] + eps[i] * __expf(0.5f * lv[i]);
        zrbf[i] = f2bf(v);
    }
}

// ---------- WMMA GEMM ----------
// Cout[M,Nn] = A[M,K] @ Bt[Nn,K]^T   (A row-major bf16, B stored transposed/K-contiguous)
// mode 0: plain fp32 store; mode 1: +bias[n]; mode 2: +bias[n] then sigmoid
// One wave -> one 16x16 tile. 8 waves/block -> 128 rows per block.
// Wave32 fragment layout per CDNA5 ISA 7.12.2:
//   A 16x32 bf16: lane half (lane>>4) selects K-halves {0..7,16..23} vs {8..15,24..31}
//   B 32x16 bf16: same K split as A: lane half selects {0..7,16..23} vs {8..15,24..31}
//   C/D 16x16 f32: vgpr r -> row r (+8 for upper half-lanes), col = lane&15
__global__ __launch_bounds__(256)
void wmma_gemm_bf16(const __bf16* __restrict__ A, const __bf16* __restrict__ Bt,
                    float* __restrict__ Cout, const float* __restrict__ bias,
                    int M, int Nn, int K, int mode) {
    const int lane = threadIdx.x & 31;
    const int wave = threadIdx.x >> 5;
    const int tm   = blockIdx.y * 8 + wave;   // 16-row tile index
    const int tn   = blockIdx.x;              // 16-col tile index
    const int half = lane >> 4;
    const int l15  = lane & 15;
    const int m = tm * 16 + l15;
    const int n = tn * 16 + l15;

    const __bf16* arow = A  + (size_t)m * K;
    const __bf16* brow = Bt + (size_t)n * K + half * 8;

    v8f acc = {};
    for (int k0 = 0; k0 < K; k0 += 32) {
        v8bf alo = *(const v8bf*)(arow + k0 + half * 8);        // K: half*8 .. half*8+7
        v8bf ahi = *(const v8bf*)(arow + k0 + 16 + half * 8);   // K: 16+half*8 ..
        v16bf a;
#pragma unroll
        for (int j = 0; j < 8; ++j) { a[j] = alo[j]; a[j + 8] = ahi[j]; }
        v8bf blo = *(const v8bf*)(brow + k0);
        v8bf bhi = *(const v8bf*)(brow + k0 + 16);
        v16bf b;
#pragma unroll
        for (int j = 0; j < 8; ++j) { b[j] = blo[j]; b[j + 8] = bhi[j]; }
        acc = __builtin_amdgcn_wmma_f32_16x16x32_bf16(
                  false, a, false, b, (short)0, acc, false, false);
    }

#pragma unroll
    for (int r = 0; r < 8; ++r) {
        int mo = tm * 16 + half * 8 + r;
        float v = acc[r];
        if (mode >= 1) v += bias[n];
        if (mode == 2) v = 1.0f / (1.0f + __expf(-v));
        Cout[(size_t)mo * Nn + n] = v;
    }
}

// ---------- launch ----------
extern "C" void kernel_launch(void* const* d_in, const int* in_sizes, int n_in,
                              void* d_out, int out_size, void* d_ws, size_t ws_size,
                              hipStream_t stream) {
    const float*     x    = (const float*)d_in[0];
    const long long* edge = (const long long*)d_in[1];
    const float*     eps  = (const float*)d_in[2];
    const float*     W1   = (const float*)d_in[3];
    const float*     b1   = (const float*)d_in[4];
    const float*     W2   = (const float*)d_in[5];
    const float*     b2   = (const float*)d_in[6];
    const float*     Wmu  = (const float*)d_in[7];
    const float*     bmu  = (const float*)d_in[8];
    const float*     Wlv  = (const float*)d_in[9];
    const float*     blv  = (const float*)d_in[10];
    const long long* src  = edge;
    const long long* dst  = edge + E_EDGES;

    // workspace carve-out (256B aligned blocks)
    char* ws = (char*)d_ws;
    size_t off = 0;
    auto carve = [&](size_t bytes) -> void* {
        void* p = ws + off;
        off += (bytes + 255) & ~(size_t)255;
        return p;
    };
    float*  deg  = (float*) carve((size_t)N_NODES * 4);
    float*  dinv = (float*) carve((size_t)N_NODES * 4);
    __bf16* Xbf  = (__bf16*)carve((size_t)N_NODES * F_DIM * 2);
    __bf16* W1t  = (__bf16*)carve((size_t)H_DIM * F_DIM * 2);
    float*  hw   = (float*) carve((size_t)N_NODES * H_DIM * 4);
    float*  hagg = (float*) carve((size_t)N_NODES * H_DIM * 4);
    __bf16* hbf  = (__bf16*)carve((size_t)N_NODES * H_DIM * 2);
    __bf16* W2t  = (__bf16*)carve((size_t)O_DIM * H_DIM * 2);
    float*  zw   = (float*) carve((size_t)N_NODES * O_DIM * 4);
    float*  zagg = (float*) carve((size_t)N_NODES * O_DIM * 4);
    __bf16* zbf  = (__bf16*)carve((size_t)N_NODES * O_DIM * 2);
    __bf16* Wmut = (__bf16*)carve((size_t)C_DIM * O_DIM * 2);
    __bf16* Wlvt = (__bf16*)carve((size_t)C_DIM * O_DIM * 2);
    __bf16* zrbf = (__bf16*)carve((size_t)N_NODES * C_DIM * 2);

    float* outp   = (float*)d_out;
    float* adj    = outp;                                        // [N,N]
    float* mu_out = outp + (size_t)N_NODES * N_NODES;            // [N,C]
    float* lv_out = mu_out + (size_t)N_NODES * C_DIM;            // [N,C]

    const int T = 256;
    auto blocks = [](long long n) { return (unsigned)((n + 255) / 256); };

    // degrees (self-loop = 1) and deg^{-1/2}
    fill_f32<<<blocks(N_NODES), T, 0, stream>>>(deg, 1.0f, N_NODES);
    deg_count<<<blocks(E_EDGES), T, 0, stream>>>(dst, deg, E_EDGES);
    rsqrt_k<<<blocks(N_NODES), T, 0, stream>>>(deg, dinv, N_NODES);

    // layer 1: hw = X @ W1
    cvt_bf16<<<blocks((long long)N_NODES * F_DIM), T, 0, stream>>>(x, Xbf, N_NODES * F_DIM);
    cvt_t_bf16<<<blocks((long long)F_DIM * H_DIM), T, 0, stream>>>(W1, W1t, F_DIM, H_DIM);
    wmma_gemm_bf16<<<dim3(H_DIM / 16, N_NODES / 128), T, 0, stream>>>(
        Xbf, W1t, hw, nullptr, N_NODES, H_DIM, F_DIM, 0);

    // aggregate + self-loop + bias + relu
    fill_f32<<<blocks((long long)N_NODES * H_DIM), T, 0, stream>>>(hagg, 0.0f, N_NODES * H_DIM);
    scatter_add<<<blocks((long long)E_EDGES * (H_DIM / 4)), T, 0, stream>>>(
        src, dst, dinv, hw, hagg, H_DIM, E_EDGES);
    finalize_k<<<blocks((long long)N_NODES * H_DIM), T, 0, stream>>>(
        hagg, hw, dinv, b1, hagg, H_DIM, N_NODES * H_DIM, 1);

    // layer 2: zw = h @ W2
    cvt_bf16<<<blocks((long long)N_NODES * H_DIM), T, 0, stream>>>(hagg, hbf, N_NODES * H_DIM);
    cvt_t_bf16<<<blocks((long long)H_DIM * O_DIM), T, 0, stream>>>(W2, W2t, H_DIM, O_DIM);
    wmma_gemm_bf16<<<dim3(O_DIM / 16, N_NODES / 128), T, 0, stream>>>(
        hbf, W2t, zw, nullptr, N_NODES, O_DIM, H_DIM, 0);

    fill_f32<<<blocks((long long)N_NODES * O_DIM), T, 0, stream>>>(zagg, 0.0f, N_NODES * O_DIM);
    scatter_add<<<blocks((long long)E_EDGES * (O_DIM / 4)), T, 0, stream>>>(
        src, dst, dinv, zw, zagg, O_DIM, E_EDGES);
    finalize_k<<<blocks((long long)N_NODES * O_DIM), T, 0, stream>>>(
        zagg, zw, dinv, b2, zagg, O_DIM, N_NODES * O_DIM, 0);

    // latent heads: mu / logvar straight into d_out
    cvt_bf16<<<blocks((long long)N_NODES * O_DIM), T, 0, stream>>>(zagg, zbf, N_NODES * O_DIM);
    cvt_t_bf16<<<blocks((long long)O_DIM * C_DIM), T, 0, stream>>>(Wmu, Wmut, O_DIM, C_DIM);
    cvt_t_bf16<<<blocks((long long)O_DIM * C_DIM), T, 0, stream>>>(Wlv, Wlvt, O_DIM, C_DIM);
    wmma_gemm_bf16<<<dim3(C_DIM / 16, N_NODES / 128), T, 0, stream>>>(
        zbf, Wmut, mu_out, bmu, N_NODES, C_DIM, O_DIM, 1);
    wmma_gemm_bf16<<<dim3(C_DIM / 16, N_NODES / 128), T, 0, stream>>>(
        zbf, Wlvt, lv_out, blv, N_NODES, C_DIM, O_DIM, 1);

    // reparameterize -> bf16 zr; row-major zr IS the K-contiguous transposed B
    reparam_k<<<blocks((long long)N_NODES * C_DIM), T, 0, stream>>>(
        mu_out, lv_out, eps, zrbf, N_NODES * C_DIM);

    // adj = sigmoid(zr @ zr^T)  [8192 x 8192], store-bandwidth bound
    wmma_gemm_bf16<<<dim3(N_NODES / 16, N_NODES / 128), T, 0, stream>>>(
        zrbf, zrbf, adj, nullptr, N_NODES, N_NODES, C_DIM, 2);
}
